// SiamMAE_68341519614425
// MI455X (gfx1250) — matrix operand order repaired and checked by reading, not compiled, not run on hardware
//
#include <hip/hip_runtime.h>
#include <math.h>

typedef __attribute__((ext_vector_type(16))) __bf16 v16bf;
typedef __attribute__((ext_vector_type(2)))  __bf16 bf16x2;
typedef __attribute__((ext_vector_type(8)))  float  v8f;

#define WMMA_BF16(a, b, c) \
    __builtin_amdgcn_wmma_f32_16x16x32_bf16(false, (a), false, (b), (short)0, (c), false, false)

// ---------------------------------------------------------------------------
// GEMM: C[M,N] = op(A[M,K] @ W[K,N] + bias) (+ res). fp32 mem, bf16 WMMA math.
// 256 threads = 8 waves; WG tile 64x128; wave tile 16x64 (4 16x16 accums).
// W is staged TRANSPOSED in LDS so B-fragment reads are row-contiguous.
// OP: 0 = none, 1 = exact GELU. HAS_RES adds res[M,N] after op.
// ---------------------------------------------------------------------------
template<int OP, bool HAS_RES>
__global__ __launch_bounds__(256) void gemm_bf16_kernel(
    const float* __restrict__ A, const float* __restrict__ W,
    const float* __restrict__ bias, const float* __restrict__ res,
    float* __restrict__ C, int M, int N, int K)
{
    __shared__ __bf16 sA [64][34];    // [m][k]
    __shared__ __bf16 sBt[128][34];   // [n][k]  (transposed)

    const int tid  = threadIdx.x;
    const int lane = tid & 31;
    const int wave = tid >> 5;
    const int hi   = lane >> 4;
    const int l15  = lane & 15;
    const int wm   = wave & 3;    // row tile 0..3 (16 rows each)
    const int wn   = wave >> 2;   // col half 0..1 (64 cols each)
    const int m0   = blockIdx.y * 64;
    const int n0   = blockIdx.x * 128;

    v8f acc[4] = {{}, {}, {}, {}};

    for (int k0 = 0; k0 < K; k0 += 32) {
        // A tile 64x32: 1024 bf16-pairs along k, 4 per thread, float2 loads
        #pragma unroll
        for (int i = 0; i < 4; ++i) {
            int p = tid + i * 256;
            int r = p >> 4, c2 = (p & 15) * 2;
            float2 f = *reinterpret_cast<const float2*>(
                &A[(size_t)(m0 + r) * K + k0 + c2]);
            bf16x2 hh; hh[0] = (__bf16)f.x; hh[1] = (__bf16)f.y;
            *reinterpret_cast<bf16x2*>(&sA[r][c2]) = hh;
        }
        // W tile 32x128 -> sBt[n][k]: 2048 pairs along k, 8 per thread
        #pragma unroll
        for (int i = 0; i < 8; ++i) {
            int p = tid + i * 256;
            int kp = p >> 7, c = p & 127;
            float f0 = W[(size_t)(k0 + 2 * kp)     * N + n0 + c];
            float f1 = W[(size_t)(k0 + 2 * kp + 1) * N + n0 + c];
            bf16x2 hh; hh[0] = (__bf16)f0; hh[1] = (__bf16)f1;
            *reinterpret_cast<bf16x2*>(&sBt[c][2 * kp]) = hh;
        }
        __syncthreads();

        v16bf aF;
        #pragma unroll
        for (int j = 0; j < 16; ++j)
            aF[j] = sA[16 * wm + l15][8 * hi + (j & 7) + ((j & 8) << 1)];

        #pragma unroll
        for (int t = 0; t < 4; ++t) {
            v16bf bF;
            #pragma unroll
            for (int j = 0; j < 16; ++j)
                bF[j] = sBt[64 * wn + 16 * t + l15][8 * hi + (j & 7) + ((j & 8) << 1)];
            acc[t] = WMMA_BF16(aF, bF, acc[t]);
        }
        __syncthreads();
    }

    #pragma unroll
    for (int t = 0; t < 4; ++t) {
        const int ct = n0 + 64 * wn + 16 * t + l15;
        const float bia = bias[ct];
        #pragma unroll
        for (int r = 0; r < 8; ++r) {
            int grow = m0 + 16 * wm + r + 8 * hi;
            float v = acc[t][r] + bia;
            if (OP == 1)
                v = 0.5f * v * (1.0f + erff(v * 0.70710678118654752f));
            if (HAS_RES)
                v += res[(size_t)grow * N + ct];
            C[(size_t)grow * N + ct] = v;
        }
    }
}

// ---------------------------------------------------------------------------
// Flash attention, head_dim = 32, NO 1/sqrt(d) scaling (faithful to source).
// grid = (N/128, B*H); 256 threads = 8 waves; wave owns 16 query rows.
// V staged TRANSPOSED so the V B-fragment reads are row-contiguous.
// Output written [B,H,N,32]-contiguous (reproduces the reference's merge bug).
// ---------------------------------------------------------------------------
__global__ __launch_bounds__(256) void flash_attn_kernel(
    const float* __restrict__ QB, const float* __restrict__ KB,
    const float* __restrict__ VB,
    int ldq, int ldk, int qoff, int koff, int voff,
    float* __restrict__ O, int N, int H)
{
    __shared__ __bf16 sK [32][34];       // [key][dim]
    __shared__ __bf16 sVt[32][34];       // [dim][key]  (transposed)
    __shared__ __bf16 sP[8][16][34];     // per-wave exp(S) tile [qrow][key]

    const int tid  = threadIdx.x;
    const int lane = tid & 31;
    const int wave = tid >> 5;
    const int hi   = lane >> 4;
    const int l15  = lane & 15;
    const int bh   = blockIdx.y;
    const int b    = bh / H;
    const int h    = bh % H;
    const size_t rowbase = (size_t)b * N;
    const int qc = qoff + h * 32;
    const int kc = koff + h * 32;
    const int vc = voff + h * 32;
    const int q0 = blockIdx.x * 128 + wave * 16;

    // Q fragment (A-matrix 16x32 layout): 4 x float4 per lane
    v16bf aQ;
    {
        const float* qp = QB + (rowbase + q0 + l15) * ldq + qc + 8 * hi;
        float4 f0 = *reinterpret_cast<const float4*>(qp);
        float4 f1 = *reinterpret_cast<const float4*>(qp + 4);
        float4 f2 = *reinterpret_cast<const float4*>(qp + 16);
        float4 f3 = *reinterpret_cast<const float4*>(qp + 20);
        aQ[0]=(__bf16)f0.x; aQ[1]=(__bf16)f0.y; aQ[2]=(__bf16)f0.z; aQ[3]=(__bf16)f0.w;
        aQ[4]=(__bf16)f1.x; aQ[5]=(__bf16)f1.y; aQ[6]=(__bf16)f1.z; aQ[7]=(__bf16)f1.w;
        aQ[8]=(__bf16)f2.x; aQ[9]=(__bf16)f2.y; aQ[10]=(__bf16)f2.z; aQ[11]=(__bf16)f2.w;
        aQ[12]=(__bf16)f3.x; aQ[13]=(__bf16)f3.y; aQ[14]=(__bf16)f3.z; aQ[15]=(__bf16)f3.w;
    }

    float mrun[8], lrun[8];
    #pragma unroll
    for (int r = 0; r < 8; ++r) { mrun[r] = -1e30f; lrun[r] = 0.0f; }
    v8f o0 = {}; v8f o1 = {};

    for (int kb = 0; kb < N; kb += 32) {
        // K tile 32x32 row-major: 512 pairs along dim, 2 per thread
        #pragma unroll
        for (int i = 0; i < 2; ++i) {
            int p = tid + i * 256;
            int r = p >> 4, c2 = (p & 15) * 2;
            float2 f = *reinterpret_cast<const float2*>(
                &KB[(rowbase + kb + r) * ldk + kc + c2]);
            bf16x2 hh; hh[0] = (__bf16)f.x; hh[1] = (__bf16)f.y;
            *reinterpret_cast<bf16x2*>(&sK[r][c2]) = hh;
        }
        // V tile 32x32 -> sVt[dim][key]: 512 pairs along key, 2 per thread
        #pragma unroll
        for (int i = 0; i < 2; ++i) {
            int p = tid + i * 256;
            int kp = p >> 5, c = p & 31;
            float f0 = VB[(rowbase + kb + 2 * kp)     * ldk + vc + c];
            float f1 = VB[(rowbase + kb + 2 * kp + 1) * ldk + vc + c];
            bf16x2 hh; hh[0] = (__bf16)f0; hh[1] = (__bf16)f1;
            *reinterpret_cast<bf16x2*>(&sVt[c][2 * kp]) = hh;
        }
        __syncthreads();

        // S = Q @ K^T : B[kdim][key] = K[key][kdim] (row-contiguous in sK)
        v16bf bK0, bK1;
        #pragma unroll
        for (int j = 0; j < 16; ++j) {
            int kk = 8 * hi + (j & 7) + ((j & 8) << 1);
            bK0[j] = sK[l15][kk];
            bK1[j] = sK[16 + l15][kk];
        }
        v8f z = {};
        v8f s0 = WMMA_BF16(aQ, bK0, z);
        v8f s1 = WMMA_BF16(aQ, bK1, z);

        // ---- online softmax, level-major butterflies over the 16-lane half
        float tm[8];
        #pragma unroll
        for (int r = 0; r < 8; ++r) tm[r] = fmaxf(s0[r], s1[r]);
        #pragma unroll
        for (int off = 1; off < 16; off <<= 1) {
            float t2[8];
            #pragma unroll
            for (int r = 0; r < 8; ++r) t2[r] = __shfl_xor(tm[r], off, 32);
            #pragma unroll
            for (int r = 0; r < 8; ++r) tm[r] = fmaxf(tm[r], t2[r]);
        }
        float sc[8], p0[8], p1[8], rs[8];
        #pragma unroll
        for (int r = 0; r < 8; ++r) {
            float mn = fmaxf(mrun[r], tm[r]);
            sc[r] = __expf(mrun[r] - mn);
            p0[r] = __expf(s0[r] - mn);
            p1[r] = __expf(s1[r] - mn);
            rs[r] = p0[r] + p1[r];
            mrun[r] = mn;
        }
        #pragma unroll
        for (int off = 1; off < 16; off <<= 1) {
            float t2[8];
            #pragma unroll
            for (int r = 0; r < 8; ++r) t2[r] = __shfl_xor(rs[r], off, 32);
            #pragma unroll
            for (int r = 0; r < 8; ++r) rs[r] += t2[r];
        }
        #pragma unroll
        for (int r = 0; r < 8; ++r) {
            lrun[r] = lrun[r] * sc[r] + rs[r];
            o0[r] *= sc[r];
            o1[r] *= sc[r];
            sP[wave][r + 8 * hi][l15]      = (__bf16)p0[r];
            sP[wave][r + 8 * hi][16 + l15] = (__bf16)p1[r];
        }
        __syncthreads();

        // O += P @ V  (both fragments row-contiguous in LDS)
        v16bf aP, bV0, bV1;
        #pragma unroll
        for (int j = 0; j < 16; ++j) {
            int ka = 8 * hi + (j & 7) + ((j & 8) << 1);
            aP[j] = sP[wave][l15][ka];
            int kk = 8 * hi + (j & 7) + ((j & 8) << 1);
            bV0[j] = sVt[l15][kk];
            bV1[j] = sVt[16 + l15][kk];
        }
        o0 = WMMA_BF16(aP, bV0, o0);
        o1 = WMMA_BF16(aP, bV1, o1);
        __syncthreads();
    }

    #pragma unroll
    for (int r = 0; r < 8; ++r) {
        float inv = 1.0f / lrun[r];
        int row = q0 + r + 8 * hi;
        size_t base = ((size_t)(b * H + h) * N + row) * 32;  // [B,H,N,32] (merge bug)
        O[base + l15]      = o0[r] * inv;
        O[base + 16 + l15] = o1[r] * inv;
    }
}

// ---------------------------------------------------------------------------
__global__ void add_kernel(const float4* __restrict__ a, const float4* __restrict__ b,
                           float4* __restrict__ y, size_t n4)
{
    size_t i = (size_t)blockIdx.x * blockDim.x + threadIdx.x;
    size_t stride = (size_t)gridDim.x * blockDim.x;
    for (; i < n4; i += stride) {
        float4 va = a[i], vb = b[i];
        float4 vy;
        vy.x = va.x + vb.x; vy.y = va.y + vb.y;
        vy.z = va.z + vb.z; vy.w = va.w + vb.w;
        y[i] = vy;
    }
}

// LayerNorm, one wave per row (D=512), float4 vectorized
__global__ __launch_bounds__(256) void ln_kernel(const float* __restrict__ X,
    const float* __restrict__ g, const float* __restrict__ bta,
    float* __restrict__ Y, int D)
{
    const int lane = threadIdx.x & 31;
    const int wave = threadIdx.x >> 5;
    const size_t row = (size_t)blockIdx.x * 8 + wave;
    const float4* x4 = reinterpret_cast<const float4*>(X + row * D);
    const int D4 = D >> 2;
    float s = 0.0f, s2 = 0.0f;
    for (int i = lane; i < D4; i += 32) {
        float4 v = x4[i];
        s  += v.x + v.y + v.z + v.w;
        s2 += v.x * v.x + v.y * v.y + v.z * v.z + v.w * v.w;
    }
    #pragma unroll
    for (int off = 1; off < 32; off <<= 1) {
        s  += __shfl_xor(s,  off, 32);
        s2 += __shfl_xor(s2, off, 32);
    }
    float mu   = s / D;
    float var  = s2 / D - mu * mu;
    float rstd = rsqrtf(var + 1e-5f);
    const float4* g4 = reinterpret_cast<const float4*>(g);
    const float4* b4 = reinterpret_cast<const float4*>(bta);
    float4* y4 = reinterpret_cast<float4*>(Y + row * D);
    for (int i = lane; i < D4; i += 32) {
        float4 v = x4[i], gg = g4[i], bb = b4[i], o;
        o.x = (v.x - mu) * rstd * gg.x + bb.x;
        o.y = (v.y - mu) * rstd * gg.y + bb.y;
        o.z = (v.z - mu) * rstd * gg.z + bb.z;
        o.w = (v.w - mu) * rstd * gg.w + bb.w;
        y4[i] = o;
    }
}

// ---------------------------------------------------------------------------
extern "C" void kernel_launch(void* const* d_in, const int* in_sizes, int n_in,
                              void* d_out, int out_size, void* d_ws, size_t ws_size,
                              hipStream_t stream)
{
    const float* x1    = (const float*)d_in[0];
    const float* x2    = (const float*)d_in[1];
    const float* qkv_w = (const float*)d_in[2];
    const float* qkv_b = (const float*)d_in[3];
    const float* q_w   = (const float*)d_in[4];
    const float* q_b   = (const float*)d_in[5];
    const float* kv_w  = (const float*)d_in[6];
    const float* kv_b  = (const float*)d_in[7];
    const float* ng    = (const float*)d_in[8];
    const float* nb    = (const float*)d_in[9];
    const float* fc1_w = (const float*)d_in[10];
    const float* fc1_b = (const float*)d_in[11];
    const float* fc2_w = (const float*)d_in[12];
    const float* fc2_b = (const float*)d_in[13];

    const int B = 4, N = 2048, D = 512, H = 16, HD = 2048;
    const int M = B * N;                       // 8192
    const size_t MD = (size_t)M * D;           // 4M floats

    float* ws = (float*)d_ws;
    // Region 0 (16M floats): q[0..MD) + kv[MD..3MD)  ->  qkv (3MD)  ->  h (4MD)
    float* bufQ   = ws;
    float* bufKV  = ws + MD;
    float* bufQKV = ws;
    float* bufH   = ws;
    float* bufC   = ws + (size_t)16 * 1024 * 1024;   // attn out, MD
    float* bufX   = bufC + MD;                       // residual stream
    float* bufN   = bufX + MD;                       // normed
    (void)in_sizes; (void)n_in; (void)out_size; (void)ws_size;

    dim3 blk(256);

    // 1. q = x2 @ q_w + q_b
    gemm_bf16_kernel<0, false><<<dim3(D / 128, M / 64), blk, 0, stream>>>(
        x2, q_w, q_b, nullptr, bufQ, M, D, D);
    // 2. kv = x1 @ kv_w + kv_b
    gemm_bf16_kernel<0, false><<<dim3(2 * D / 128, M / 64), blk, 0, stream>>>(
        x1, kv_w, kv_b, nullptr, bufKV, M, 2 * D, D);
    // 3. cross attention -> bufC ([B,H,N,32] layout == buggy merge)
    flash_attn_kernel<<<dim3(N / 128, B * H), blk, 0, stream>>>(
        bufQ, bufKV, bufKV, D, 2 * D, 0, 0, D, bufC, N, H);
    // 4. x = x2 + attn
    add_kernel<<<1024, 256, 0, stream>>>((const float4*)x2, (const float4*)bufC,
                                         (float4*)bufX, MD / 4);
    // 5. n1 = LN(x)
    ln_kernel<<<M / 8, 256, 0, stream>>>(bufX, ng, nb, bufN, D);
    // 6. qkv = n1 @ qkv_w + qkv_b
    gemm_bf16_kernel<0, false><<<dim3(3 * D / 128, M / 64), blk, 0, stream>>>(
        bufN, qkv_w, qkv_b, nullptr, bufQKV, M, 3 * D, D);
    // 7. self attention -> bufC
    flash_attn_kernel<<<dim3(N / 128, B * H), blk, 0, stream>>>(
        bufQKV, bufQKV, bufQKV, 3 * D, 3 * D, 0, D, 2 * D, bufC, N, H);
    // 8. x = n1 + attn
    add_kernel<<<1024, 256, 0, stream>>>((const float4*)bufN, (const float4*)bufC,
                                         (float4*)bufX, MD / 4);
    // 9. n2 = LN(x)
    ln_kernel<<<M / 8, 256, 0, stream>>>(bufX, ng, nb, bufN, D);
    // 10. h = gelu(n2 @ fc1_w + fc1_b)
    gemm_bf16_kernel<1, false><<<dim3(HD / 128, M / 64), blk, 0, stream>>>(
        bufN, fc1_w, fc1_b, nullptr, bufH, M, HD, D);
    // 11. out = n2 + h @ fc2_w + fc2_b
    gemm_bf16_kernel<0, true><<<dim3(D / 128, M / 64), blk, 0, stream>>>(
        bufH, fc2_w, fc2_b, bufN, (float*)d_out, M, D, HD);
}
